// KHAOS_KAN_94489281314
// MI455X (gfx1250) — matrix operand order repaired and checked by reading, not compiled, not run on hardware
//
#include <hip/hip_runtime.h>
#include <math.h>

#define BATCH     8192
#define IN_F      1024
#define OUT_F     1024
#define GK        8                  // grid_size + spline_order
#define K_TOTAL   (IN_F + IN_F * GK) // 9216
#define BM        128
#define BN        128
#define BK        32
#define NK        (K_TOTAL / BK)     // 288
#define LDSS      56                 // padded LDS row stride in bf16 elements (112 B)

typedef __attribute__((ext_vector_type(16))) __bf16 v16bf;
typedef __attribute__((ext_vector_type(8)))  float  v8f;

#if defined(__has_builtin)
#if __has_builtin(__builtin_amdgcn_tensor_load_to_lds) && \
    __has_builtin(__builtin_amdgcn_s_wait_tensorcnt)
#define USE_TDM 1
#endif
#endif

// f32 -> bf16 with round-to-nearest-even (bit-exact, no __bf16 scalar math needed)
__device__ __forceinline__ unsigned short f2bf(float f) {
    unsigned int u = __float_as_uint(f);
    u += 0x7FFFu + ((u >> 16) & 1u);
    return (unsigned short)(u >> 16);
}

// 8 uniform cubic B-spline basis values (bf16-packed in a uint4) for one x.
// Closed form == de Boor on uniform knots t_j = -2.2 + 0.4 j; only 4 of the
// 8 are nonzero. Branch-free: the 4 weights are packed into 64 bits and
// funnel-shifted into position inside a 128-bit window (no arrays, no scratch).
__device__ __forceinline__ uint4 bspline8v(float xv) {
    float s  = (xv + 2.2f) * 2.5f;           // (x - t0) / h
    int   j0 = (int)floorf(s);
    j0 = j0 < 3 ? 3 : (j0 > 10 ? 10 : j0);
    float u  = s - (float)j0;                // in [0,1)
    float um = 1.0f - u;
    float u2 = u * u, u3 = u2 * u;
    float w0 = um * um * um * (1.0f / 6.0f);
    float w1 = (3.0f * u3 - 6.0f * u2 + 4.0f) * (1.0f / 6.0f);
    float w2 = (-3.0f * u3 + 3.0f * u2 + 3.0f * u + 1.0f) * (1.0f / 6.0f);
    float w3 = u3 * (1.0f / 6.0f);

    unsigned long long wpk =
          (unsigned long long)f2bf(w0)
        | ((unsigned long long)f2bf(w1) << 16)
        | ((unsigned long long)f2bf(w2) << 32)
        | ((unsigned long long)f2bf(w3) << 48);

    const unsigned sh = (unsigned)(j0 - 3) * 16u;   // 0..112 bits
    unsigned long long lo, hi;
    if (sh == 0)       { lo = wpk;             hi = 0ull; }
    else if (sh < 64u) { lo = wpk << sh;       hi = wpk >> (64u - sh); }
    else               { lo = 0ull;            hi = wpk << (sh - 64u); }
    // bits shifted past bit 127 vanish == bases truncated at d=7 (de Boor edge)

    uint4 r;
    r.x = (unsigned)lo;
    r.y = (unsigned)(lo >> 32);
    r.z = (unsigned)hi;
    r.w = (unsigned)(hi >> 32);
    return r;
}

#ifdef USE_TDM
typedef __attribute__((ext_vector_type(4))) unsigned int v4u;
typedef __attribute__((ext_vector_type(8))) int          v8i_t;
typedef __attribute__((ext_vector_type(4))) int          v4i_t;

// LDS byte offset of a __shared__ object (generic -> AS3 -> integer)
__device__ __forceinline__ unsigned lds_off(const void* p) {
    return (unsigned)(unsigned long long)
        (__attribute__((address_space(3))) const void*)p;
}

// TDM: DMA a 128x32 bf16 tile (row stride K_TOTAL elems in memory) into LDS,
// padding each 64-B row to 112 B (pad_interval = 16 DW, pad_amount = 12 DW)
// to match the LDSS=56 fragment layout. D# per CDNA5 ISA §8.3/8.4.
__device__ __forceinline__ void tdm_load_b_tile(const unsigned short* gsrc,
                                                unsigned lds_byte) {
    unsigned long long ga = (unsigned long long)(const void*)gsrc;
    v4u g0;
    g0[0] = 1u;                                   // count=1, user desc
    g0[1] = lds_byte;                             // lds_addr [63:32]
    g0[2] = (unsigned)ga;                         // global_addr lo
    g0[3] = (unsigned)(ga >> 32) | (2u << 30);    // global_addr hi | type=2
    v8i_t g1;
    g1[0] = (int)((1u << 16)      // data_size = 2 bytes
                | (1u << 20)      // pad_enable
                | (3u << 22)      // pad_interval: 16 DWORDs (64 B)
                | (11u << 25));   // pad_amount:  12 DWORDs (48 B)
    g1[1] = (int)((K_TOTAL & 0xFFFFu) << 16);     // tensor_dim0 lo16 @ [63:48]
    g1[2] = (int)(128u << 16);                    // dim0 hi=0 | tensor_dim1=128
    g1[3] = (int)(32u << 16);                     // dim1 hi=0 | tile_dim0=32
    g1[4] = (int)128u;                            // tile_dim1=128 | tile_dim2=0
    g1[5] = (int)K_TOTAL;                         // tensor_dim0_stride = 9216
    g1[6] = 0;
    g1[7] = 0;
    v4i_t z4 = {0, 0, 0, 0};                      // groups 2/3 unused (2D)
    v8i_t z8 = {0, 0, 0, 0, 0, 0, 0, 0};          // extra group (6-arg form)
    __builtin_amdgcn_tensor_load_to_lds(g0, g1, z4, z4, z8, 0);
}
#endif

// ---------------------------------------------------------------------------
// Prep: pack [ base_weight | spline_weight * scaler ] as bf16 (OUT_F x K_TOTAL)
// ---------------------------------------------------------------------------
__global__ void kan_pack_weights(const float* __restrict__ bw,
                                 const float* __restrict__ sw,
                                 const float* __restrict__ ss,
                                 unsigned short* __restrict__ wp) {
    int idx = blockIdx.x * blockDim.x + threadIdx.x;
    if (idx >= OUT_F * K_TOTAL) return;
    int o = idx / K_TOTAL;
    int k = idx - o * K_TOTAL;
    float v;
    if (k < IN_F) {
        v = bw[o * IN_F + k];
    } else {
        int kp = k - IN_F;
        int i = kp >> 3;
        int d = kp & 7;
        v = sw[((size_t)(o * IN_F + i)) * GK + d] * ss[o * IN_F + i];
    }
    wp[idx] = f2bf(v);
}

// ---------------------------------------------------------------------------
// Fused double-buffered GEMM: out[b,o] = sum_k A[b,k] * Wp[o,k]
//   A[b, 0:1024]          = silu(x[b, :])        (computed in-register)
//   A[b, 1024 + 8*i + d]  = cubic B-spline basis (computed in-register)
//   B tile: TDM DMA global->LDS when available, else vector path
// ---------------------------------------------------------------------------
__global__ __launch_bounds__(256) void kan_wmma_gemm(
    const float* __restrict__ x,
    const unsigned short* __restrict__ wp,
    float* __restrict__ out)
{
    __shared__ __align__(16) unsigned short As[2][BM * LDSS];
    __shared__ __align__(16) unsigned short Bs[2][BN * LDSS];

    const int t    = threadIdx.x;
    const int lane = t & 31;
    const int wave = t >> 5;      // 0..7
    const int wm   = wave >> 2;   // 0..1 -> 64-row slab
    const int wn   = wave & 3;    // 0..3 -> 32-col slab
    const int bm   = (int)blockIdx.x * BM;
    const int bn   = (int)blockIdx.y * BN;

    const int row2 = t >> 1;         // 0..127: tile row handled by this thread
    const int cb   = (t & 1) << 4;   // 0 or 16: K-column half
    const int mS   = t >> 2;         // spline task 0 row (0..63), task1 = +64
    const int fiS  = t & 3;          // spline feature-in-chunk

    v8f acc[4][2];
    #pragma unroll
    for (int mi = 0; mi < 4; ++mi)
        #pragma unroll
        for (int ni = 0; ni < 2; ++ni)
            #pragma unroll
            for (int e = 0; e < 8; ++e) acc[mi][ni][e] = 0.0f;

    // ---- prologue: tile 0 (base region) into buffer 0 ----
#ifdef USE_TDM
    if (wave == 0)
        tdm_load_b_tile(wp + (size_t)bn * K_TOTAL, lds_off(&Bs[0][0]));
#else
    {
        const unsigned short* gp = wp + (size_t)(bn + row2) * K_TOTAL + cb;
        uint4 q0 = ((const uint4*)gp)[0];
        uint4 q1 = ((const uint4*)gp)[1];
        uint4* dpB = (uint4*)(Bs[0] + row2 * LDSS + cb);
        dpB[0] = q0;
        dpB[1] = q1;
    }
#endif
    {
        const float* gx = x + (size_t)(bm + row2) * IN_F + cb;
        unsigned short tmp[16];
        #pragma unroll
        for (int j = 0; j < 16; ++j) {
            float v = gx[j];
            tmp[j] = f2bf(v / (1.0f + __expf(-v)));
        }
        uint4* dpA = (uint4*)(As[0] + row2 * LDSS + cb);
        dpA[0] = *(const uint4*)(tmp);
        dpA[1] = *(const uint4*)(tmp + 8);
    }
#ifdef USE_TDM
    if (wave == 0)
        __builtin_amdgcn_s_wait_tensorcnt((unsigned short)0);
#endif
    __syncthreads();

    for (int kt = 0; kt < NK; ++kt) {
        const int  cur      = kt & 1;
        const int  nxt      = cur ^ 1;
        const int  k0n      = (kt + 1) * BK;
        const bool hasNext  = (kt + 1) < NK;
        const bool nextBase = k0n < IN_F;

        // ---- kick off next B tile (DMA or staged loads, in flight across WMMAs)
#ifdef USE_TDM
        if (hasNext && wave == 0)
            tdm_load_b_tile(wp + (size_t)bn * K_TOTAL + k0n,
                            lds_off(&Bs[nxt][0]));
#else
        uint4 bq0, bq1;
        if (hasNext) {
            const unsigned short* gp =
                wp + (size_t)(bn + row2) * K_TOTAL + k0n + cb;
            bq0 = ((const uint4*)gp)[0];
            bq1 = ((const uint4*)gp)[1];
            if (kt + 2 < NK)
                __builtin_prefetch(gp + BK, 0, 0);   // global_prefetch_b8
        }
#endif
        // ---- stage next A tile's x loads ----
        float xbase[16];
        float xs0 = 0.0f, xs1 = 0.0f;
        if (hasNext) {
            if (nextBase) {
                const float* gx = x + (size_t)(bm + row2) * IN_F + k0n + cb;
                #pragma unroll
                for (int j = 0; j < 4; ++j) {
                    float4 f = ((const float4*)gx)[j];
                    xbase[4 * j + 0] = f.x;
                    xbase[4 * j + 1] = f.y;
                    xbase[4 * j + 2] = f.z;
                    xbase[4 * j + 3] = f.w;
                }
            } else {
                const int i0 = (k0n - IN_F) >> 3;
                xs0 = x[(size_t)(bm + mS) * IN_F + i0 + fiS];
                xs1 = x[(size_t)(bm + mS + 64) * IN_F + i0 + fiS];
            }
        }

        // ---- fragments (ISA layout) + 4x2 WMMA from buffer `cur` ----
        {
            union FragU { v16bf v; uint4 q[2]; };
            const int rA   = lane & 15;
            const int khA0 = (lane >> 4) * 8;    // A: K chunks {khA0, 16+khA0}
            const int kb0  = (lane >> 4) * 8;    // B: K chunks {kb0, 16+kb0}

            v16bf afrag[4];
            #pragma unroll
            for (int mi = 0; mi < 4; ++mi) {
                const unsigned short* p =
                    As[cur] + (wm * 64 + mi * 16 + rA) * LDSS;
                FragU f;
                f.q[0] = *(const uint4*)(p + khA0);
                f.q[1] = *(const uint4*)(p + 16 + khA0);
                afrag[mi] = f.v;
            }
            v16bf bfrag[2];
            #pragma unroll
            for (int ni = 0; ni < 2; ++ni) {
                const unsigned short* p =
                    Bs[cur] + (wn * 32 + ni * 16 + rA) * LDSS;
                FragU f;
                f.q[0] = *(const uint4*)(p + kb0);
                f.q[1] = *(const uint4*)(p + kb0 + 16);
                bfrag[ni] = f.v;
            }
            #pragma unroll
            for (int mi = 0; mi < 4; ++mi)
                #pragma unroll
                for (int ni = 0; ni < 2; ++ni)
                    acc[mi][ni] = __builtin_amdgcn_wmma_f32_16x16x32_bf16(
                        false, afrag[mi], false, bfrag[ni],
                        (short)0, acc[mi][ni], false, false);
        }

        // ---- convert + store next A tile (and B tile on fallback path) ----
        if (hasNext) {
#ifndef USE_TDM
            uint4* dpB = (uint4*)(Bs[nxt] + row2 * LDSS + cb);
            dpB[0] = bq0;
            dpB[1] = bq1;
#endif
            if (nextBase) {
                unsigned short tmp[16];
                #pragma unroll
                for (int j = 0; j < 16; ++j) {
                    float v = xbase[j];
                    tmp[j] = f2bf(v / (1.0f + __expf(-v)));
                }
                uint4* dpA = (uint4*)(As[nxt] + row2 * LDSS + cb);
                dpA[0] = *(const uint4*)(tmp);
                dpA[1] = *(const uint4*)(tmp + 8);
            } else {
                uint4 v0 = bspline8v(xs0);
                uint4 v1 = bspline8v(xs1);
                *(uint4*)(As[nxt] + mS * LDSS + fiS * 8)        = v0;
                *(uint4*)(As[nxt] + (mS + 64) * LDSS + fiS * 8) = v1;
            }
        }
#ifdef USE_TDM
        if (hasNext && wave == 0)
            __builtin_amdgcn_s_wait_tensorcnt((unsigned short)0);
#endif
        __syncthreads();   // orders: next-tile stores/DMA visible; cur reads done
    }

    // ---- store C: VGPR e -> row (e + (lane/16)*8), col lane%16 ----
    const int colC = lane & 15;
    const int rOff = (lane >> 4) * 8;
    #pragma unroll
    for (int mi = 0; mi < 4; ++mi)
        #pragma unroll
        for (int ni = 0; ni < 2; ++ni) {
            const int col   = bn + wn * 32 + ni * 16 + colC;
            const int rbase = bm + wm * 64 + mi * 16 + rOff;
            #pragma unroll
            for (int e = 0; e < 8; ++e)
                out[(size_t)(rbase + e) * OUT_F + col] = acc[mi][ni][e];
        }
}

extern "C" void kernel_launch(void* const* d_in, const int* in_sizes, int n_in,
                              void* d_out, int out_size, void* d_ws, size_t ws_size,
                              hipStream_t stream) {
    const float* x  = (const float*)d_in[0];   // (8192, 1024)
    const float* bw = (const float*)d_in[1];   // (1024, 1024)
    const float* sw = (const float*)d_in[2];   // (1024, 1024, 8)
    const float* ss = (const float*)d_in[3];   // (1024, 1024)
    // d_in[4] = grid: uniform, constants hardcoded
    unsigned short* wp = (unsigned short*)d_ws; // (1024, 9216) bf16 = 18.9 MB
    float* out = (float*)d_out;                 // (8192, 1024)

    const int total = OUT_F * K_TOTAL;
    kan_pack_weights<<<(total + 255) / 256, 256, 0, stream>>>(bw, sw, ss, wp);

    dim3 grid(BATCH / BM, OUT_F / BN);          // 64 x 8
    kan_wmma_gemm<<<grid, 256, 0, stream>>>(x, wp, out);
}
